// MeshNetV0_23605140259088
// MI455X (gfx1250) — matrix operand order repaired and checked by reading, not compiled, not run on hardware
//
#include <hip/hip_runtime.h>
#include <hip/hip_bf16.h>

// ---------------------------------------------------------------------------
// CDNA5 (gfx1250) MeshNet pipeline, round 2:
//  - all WMMA operands pre-converted to bf16 in memory (A row-major,
//    B transposed to [C,K]) so the GEMM inner loop is 4x global_load_b128
//    + 1x v_wmma_f32_16x16x32_bf16, no per-use cvt VALU storm.
//  - scatter-add stays fp32 (global_atomic_add_f32, L2-resident working set).
//  - bias+ReLU fused with fp32->bf16 repack (ushort4 stores).
//  - 320x1024 comb GEMM fused with global max-pool (atomicMax uint trick).
// ---------------------------------------------------------------------------

typedef __attribute__((ext_vector_type(16))) __bf16 v16bf;
typedef __attribute__((ext_vector_type(8)))  __bf16 v8bf;
typedef __attribute__((ext_vector_type(8)))  float  v8f;

__device__ __forceinline__ unsigned short f2bf(float f) {
  unsigned u = __float_as_uint(f);
  u += 0x7FFFu + ((u >> 16) & 1u);     // round-to-nearest-even
  return (unsigned short)(u >> 16);
}

// Two contiguous 16-byte bf16 chunks -> one 16-element WMMA fragment.
__device__ __forceinline__ v16bf load16(const __bf16* p0, const __bf16* p1) {
  v8bf lo = *(const v8bf*)p0;
  v8bf hi = *(const v8bf*)p1;
  return __builtin_shufflevector(lo, hi, 0, 1, 2, 3, 4, 5, 6, 7,
                                 8, 9, 10, 11, 12, 13, 14, 15);
}

// ---------------------------------------------------------------------------
// GEMM: Out[M,C] (fp32) = A[M,K] (bf16, row-major) @ Bt[C,K] (bf16, transposed)
// 8 waves/block, 16 rows/wave, wave loops over 16-wide column tiles.
// A frag: lane<16 -> K=k0..k0+7 & k0+16..23 ; lane>=16 -> +8 (ISA 16-bit A tbl)
// B frag: lane n-col, same K order as A frag (two 8-wide chunks in Bt row).
// ---------------------------------------------------------------------------
__global__ void __launch_bounds__(256)
wmma_gemm_bf16(const __bf16* __restrict__ A, const __bf16* __restrict__ Bt,
               float* __restrict__ Out, int M, int K, int C) {
  const int lane = threadIdx.x & 31;
  const int wave = threadIdx.x >> 5;
  const int row0 = (blockIdx.x * 8 + wave) * 16;
  if (row0 + 16 > M) return;                        // wave-uniform exit
  const int m    = lane & 15;
  const int hi8  = (lane >> 4) << 3;
  const int hi16 = (lane >> 4) << 4;
  const __bf16* arow = A + (size_t)(row0 + m) * K + hi8;
  for (int col0 = 0; col0 < C; col0 += 16) {
    const __bf16* brow = Bt + (size_t)(col0 + m) * K + hi8;
    v8f acc = {};
    for (int k0 = 0; k0 < K; k0 += 32) {
      v16bf a = load16(arow + k0, arow + k0 + 16);
      v16bf b = load16(brow + k0, brow + k0 + 16);
      acc = __builtin_amdgcn_wmma_f32_16x16x32_bf16(false, a, false, b,
                                                    (short)0, acc, false, false);
    }
    const int n  = col0 + m;
    const int mh = hi8;                             // D: M = r + 8*(lane>=16)
#pragma unroll
    for (int r = 0; r < 8; ++r)
      Out[(size_t)(row0 + mh + r) * C + n] = acc[r];
  }
}

// agg[dst] += hw[src] * w  (fp32; gather + 4x global_atomic_add_f32).
// hw + agg + edge lists fit the 192 MB L2 -> runs at L2/atomic rate.
__global__ void __launch_bounds__(256)
edge_scatter(const float* __restrict__ hw, const int* __restrict__ src,
             const int* __restrict__ dst, const float* __restrict__ w,
             float* __restrict__ agg, long long total, int C, int vshift) {
  long long gid = blockIdx.x * (long long)blockDim.x + threadIdx.x;
  if (gid >= total) return;
  const int e  = (int)(gid >> vshift);
  const int c4 = (int)(gid & ((1 << vshift) - 1)) << 2;
  const int s = src[e], d = dst[e];
  const float we = w[e];
  const float4 hv = *(const float4*)(hw + (size_t)s * C + c4);
  float* o = agg + (size_t)d * C + c4;
  atomicAdd(o + 0, hv.x * we);
  atomicAdd(o + 1, hv.y * we);
  atomicAdd(o + 2, hv.z * we);
  atomicAdd(o + 3, hv.w * we);
}

// out_bf16 = bf16(relu(agg + bias)); 4-wide (float4 in, ushort4 out).
__global__ void __launch_bounds__(256)
bias_relu_bf16(const float* __restrict__ agg, const float* __restrict__ b,
               unsigned short* __restrict__ out, long long total4, int C4) {
  long long i = blockIdx.x * (long long)blockDim.x + threadIdx.x;
  if (i >= total4) return;
  float4 v = ((const float4*)agg)[i];
  const float4 bv = ((const float4*)b)[(int)(i % C4)];
  ushort4 r;
  r.x = f2bf(fmaxf(v.x + bv.x, 0.f));
  r.y = f2bf(fmaxf(v.y + bv.y, 0.f));
  r.z = f2bf(fmaxf(v.z + bv.z, 0.f));
  r.w = f2bf(fmaxf(v.w + bv.w, 0.f));
  ((ushort4*)out)[i] = r;
}

// fp32 -> bf16, 4-wide (for the network input x).
__global__ void __launch_bounds__(256)
cvt_act(const float* __restrict__ in, unsigned short* __restrict__ out,
        long long total4) {
  long long i = blockIdx.x * (long long)blockDim.x + threadIdx.x;
  if (i >= total4) return;
  const float4 v = ((const float4*)in)[i];
  ushort4 r;
  r.x = f2bf(v.x); r.y = f2bf(v.y); r.z = f2bf(v.z); r.w = f2bf(v.w);
  ((ushort4*)out)[i] = r;
}

// W[K,C] fp32 -> Wt[C,K] bf16 (transpose so B frags are contiguous).
__global__ void __launch_bounds__(256)
cvt_w_t(const float* __restrict__ W, unsigned short* __restrict__ Wt,
        int K, int C) {
  int i = blockIdx.x * blockDim.x + threadIdx.x;
  if (i >= K * C) return;
  const int n = i / K, k = i - n * K;
  Wt[i] = f2bf(W[(size_t)k * C + n]);
}

// Concat(x1..x4)[N,320] @ Wcb[320,1024] + bcb, ReLU, fused column max-pool.
__global__ void __launch_bounds__(256)
comb_gemm_maxpool(const __bf16* __restrict__ x1b, const __bf16* __restrict__ x2b,
                  const __bf16* __restrict__ x3b, const __bf16* __restrict__ x4b,
                  const __bf16* __restrict__ Wcbt, const float* __restrict__ bcb,
                  unsigned* __restrict__ hmax, int M) {
  const int lane = threadIdx.x & 31;
  const int wave = threadIdx.x >> 5;
  const int row0 = (blockIdx.x * 8 + wave) * 16;
  if (row0 + 16 > M) return;                        // wave-uniform
  const int col0 = blockIdx.y * 16;
  const int m    = lane & 15;
  const int hi8  = (lane >> 4) << 3;
  const int hi16 = (lane >> 4) << 4;
  const __bf16* a1 = x1b + (size_t)(row0 + m) * 64 + hi8;
  const __bf16* a2 = x2b + (size_t)(row0 + m) * 64 + hi8;
  const __bf16* a3 = x3b + (size_t)(row0 + m) * 64 + hi8;
  const __bf16* a4 = x4b + (size_t)(row0 + m) * 128 + hi8;
  const __bf16* br = Wcbt + (size_t)(col0 + m) * 320 + hi8;
  v8f acc = {};
#define CMB_STEP(AP, KA, KB)                                                   \
  acc = __builtin_amdgcn_wmma_f32_16x16x32_bf16(                               \
      false, load16((AP) + (KA), (AP) + (KA) + 16),                            \
      false, load16(br + (KB), br + (KB) + 16), (short)0, acc, false, false);
  CMB_STEP(a1, 0, 0)    CMB_STEP(a1, 32, 32)       // x1: K 0..63
  CMB_STEP(a2, 0, 64)   CMB_STEP(a2, 32, 96)       // x2: K 64..127
  CMB_STEP(a3, 0, 128)  CMB_STEP(a3, 32, 160)      // x3: K 128..191
  CMB_STEP(a4, 0, 192)  CMB_STEP(a4, 32, 224)      // x4: K 192..319
  CMB_STEP(a4, 64, 256) CMB_STEP(a4, 96, 288)
#undef CMB_STEP
  const int n = col0 + m;
  const float bias = bcb[n];
  float mx = 0.f;                                   // ReLU floor
#pragma unroll
  for (int r = 0; r < 8; ++r) mx = fmaxf(mx, fmaxf(acc[r] + bias, 0.f));
  mx = fmaxf(mx, __shfl_xor(mx, 16, 32));          // fold row halves (wave32)
  if (lane < 16) atomicMax(&hmax[n], __float_as_uint(mx));  // monotone, f>=0
}

// FC head on the pooled [1,1024] vector. Single block of 512 threads.
__global__ void __launch_bounds__(512)
fc_head(const unsigned* __restrict__ hmax,
        const float* __restrict__ fc1_w, const float* __restrict__ fc1_b,
        const float* __restrict__ g1,    const float* __restrict__ be1,
        const float* __restrict__ fc2_w, const float* __restrict__ fc2_b,
        const float* __restrict__ g2,    const float* __restrict__ be2,
        const float* __restrict__ fc3_w, const float* __restrict__ fc3_b,
        float* __restrict__ out) {
  __shared__ float h[1024];
  __shared__ float h1[512];
  __shared__ float h2[256];
  const int tid = threadIdx.x;
  h[tid]       = __uint_as_float(hmax[tid]);
  h[tid + 512] = __uint_as_float(hmax[tid + 512]);
  __syncthreads();
  const float bns = rsqrtf(1.0f + 1e-5f);
  {
    float acc = fc1_b[tid];
    for (int k = 0; k < 1024; ++k) acc += fc1_w[(size_t)tid * 1024 + k] * h[k];
    h1[tid] = fmaxf(acc * (g1[tid] * bns) + be1[tid], 0.f);
  }
  __syncthreads();
  if (tid < 256) {
    float acc = fc2_b[tid];
    for (int k = 0; k < 512; ++k) acc += fc2_w[(size_t)tid * 512 + k] * h1[k];
    h2[tid] = fmaxf(acc * (g2[tid] * bns) + be2[tid], 0.f);
  }
  __syncthreads();
  if (tid < 40) {
    float acc = fc3_b[tid];
    for (int k = 0; k < 256; ++k) acc += fc3_w[(size_t)tid * 256 + k] * h2[k];
    out[tid] = acc;
  }
}

extern "C" void kernel_launch(void* const* d_in, const int* in_sizes, int n_in,
                              void* d_out, int out_size, void* d_ws, size_t ws_size,
                              hipStream_t stream) {
  const float* x    = (const float*)d_in[0];
  const int*   esrc = (const int*)  d_in[1];
  const int*   edst = (const int*)  d_in[2];
  const float* ew   = (const float*)d_in[3];
  const float* W1 = (const float*)d_in[4];  const float* b1 = (const float*)d_in[5];
  const float* W2 = (const float*)d_in[6];  const float* b2 = (const float*)d_in[7];
  const float* W3 = (const float*)d_in[8];  const float* b3 = (const float*)d_in[9];
  const float* W4 = (const float*)d_in[10]; const float* b4 = (const float*)d_in[11];
  const float* Wcb = (const float*)d_in[12]; const float* bcb = (const float*)d_in[13];
  const float* fc1_w = (const float*)d_in[14]; const float* fc1_b = (const float*)d_in[15];
  const float* g1 = (const float*)d_in[16];    const float* be1  = (const float*)d_in[17];
  const float* fc2_w = (const float*)d_in[18]; const float* fc2_b = (const float*)d_in[19];
  const float* g2 = (const float*)d_in[20];    const float* be2  = (const float*)d_in[21];
  const float* fc3_w = (const float*)d_in[22]; const float* fc3_b = (const float*)d_in[23];

  const int N = in_sizes[0] / 64;   // 100000
  const int E = in_sizes[1];        // 3200000

  // Workspace (all offsets 16B-aligned):
  //   hw fp32 [N*128] | agg fp32 [N*128] | x0b..x4b bf16 | weights bf16 | hmax
  float* hw  = (float*)d_ws;
  float* agg = hw + (size_t)N * 128;
  unsigned short* x0b = (unsigned short*)(agg + (size_t)N * 128);
  unsigned short* x1b = x0b + (size_t)N * 64;
  unsigned short* x2b = x1b + (size_t)N * 64;
  unsigned short* x3b = x2b + (size_t)N * 64;
  unsigned short* x4b = x3b + (size_t)N * 64;
  unsigned short* w1t = x4b + (size_t)N * 128;     // [64,64]
  unsigned short* w2t = w1t + 64 * 64;             // [64,64]
  unsigned short* w3t = w2t + 64 * 64;             // [64,64]
  unsigned short* w4t = w3t + 64 * 64;             // [128,64]
  unsigned short* wct = w4t + 128 * 64;            // [1024,320]
  unsigned* hmax = (unsigned*)(wct + (size_t)1024 * 320);

  // --- one-time (per launch) conversions: input + transposed bf16 weights ---
  {
    const long long t4 = (long long)N * 64 / 4;
    cvt_act<<<(int)((t4 + 255) / 256), 256, 0, stream>>>(x, x0b, t4);
    cvt_w_t<<<(64 * 64 + 255) / 256, 256, 0, stream>>>(W1, w1t, 64, 64);
    cvt_w_t<<<(64 * 64 + 255) / 256, 256, 0, stream>>>(W2, w2t, 64, 64);
    cvt_w_t<<<(64 * 64 + 255) / 256, 256, 0, stream>>>(W3, w3t, 64, 64);
    cvt_w_t<<<(64 * 128 + 255) / 256, 256, 0, stream>>>(W4, w4t, 64, 128);
    cvt_w_t<<<(320 * 1024 + 255) / 256, 256, 0, stream>>>(Wcb, wct, 320, 1024);
  }

  const int gemmBlocks = (N + 127) / 128;

  auto layer = [&](const unsigned short* inb, const unsigned short* Wt,
                   const float* b, unsigned short* outb, int C) {
    wmma_gemm_bf16<<<gemmBlocks, 256, 0, stream>>>(
        (const __bf16*)inb, (const __bf16*)Wt, hw, N, 64, C);
    hipMemsetAsync(agg, 0, (size_t)N * C * sizeof(float), stream);
    const int vshift = (C == 64) ? 4 : 5;          // float4 chunks per edge
    const long long total = (long long)E << vshift;
    edge_scatter<<<(int)((total + 255) / 256), 256, 0, stream>>>(
        hw, esrc, edst, ew, agg, total, C, vshift);
    const long long tot4 = (long long)N * C / 4;
    bias_relu_bf16<<<(int)((tot4 + 255) / 256), 256, 0, stream>>>(
        agg, b, outb, tot4, C / 4);
  };

  layer(x0b, w1t, b1, x1b, 64);
  layer(x1b, w2t, b2, x2b, 64);
  layer(x2b, w3t, b3, x3b, 64);
  layer(x3b, w4t, b4, x4b, 128);

  hipMemsetAsync(hmax, 0, 1024 * sizeof(unsigned), stream);
  comb_gemm_maxpool<<<dim3(gemmBlocks, 64), 256, 0, stream>>>(
      (const __bf16*)x1b, (const __bf16*)x2b, (const __bf16*)x3b,
      (const __bf16*)x4b, (const __bf16*)wct, bcb, hmax, N);

  fc_head<<<1, 512, 0, stream>>>(hmax, fc1_w, fc1_b, g1, be1,
                                 fc2_w, fc2_b, g2, be2, fc3_w, fc3_b,
                                 (float*)d_out);
}
